// GCN_82755429859973
// MI455X (gfx1250) — matrix operand order repaired and checked by reading, not compiled, not run on hardware
//
#include <hip/hip_runtime.h>
#include <stddef.h>

// ---------- CDNA5 WMMA types ----------
typedef __attribute__((ext_vector_type(16))) __bf16 v16bf;
typedef __attribute__((ext_vector_type(8)))  __bf16 v8bf;
typedef __attribute__((ext_vector_type(8)))  float  v8f;
typedef int v4i __attribute__((vector_size(16)));   // param type of async-LDS builtin

#define BM 128
#define BN 128
#define BK 32
#define GEMM_THREADS 256   // 8 wave32: 4 along M (32 rows) x 2 along N (64 cols)

// CDNA5 async global->LDS copy path (ASYNCcnt), if the toolchain exposes it
#if defined(__has_builtin)
#if __has_builtin(__builtin_amdgcn_global_load_async_to_lds_b128) && \
    __has_builtin(__builtin_amdgcn_s_wait_asynccnt)
#define USE_ASYNC_LDS 1
#endif
#endif
#ifndef USE_ASYNC_LDS
#define USE_ASYNC_LDS 0
#endif

// fp32 -> bf16 bits, round-to-nearest-even
static __device__ __forceinline__ unsigned short f2bf(float f) {
    unsigned int u = __float_as_uint(f);
    unsigned int r = 0x7FFFu + ((u >> 16) & 1u);
    return (unsigned short)((u + r) >> 16);
}

// Load a 16-element bf16 fragment from two 16B-aligned LDS addresses
static __device__ __forceinline__ v16bf load_frag(const unsigned short* p0,
                                                  const unsigned short* p1) {
    v8bf lo = *reinterpret_cast<const v8bf*>(p0);
    v8bf hi = *reinterpret_cast<const v8bf*>(p1);
    return __builtin_shufflevector(lo, hi, 0, 1, 2, 3, 4, 5, 6, 7,
                                            8, 9, 10, 11, 12, 13, 14, 15);
}

// Stage one 128x32 bf16 tile (G row-major, stride K) into LDS tile S.
// full==true: every row in range -> unguarded (async DMA if available).
static __device__ __forceinline__ void stage_tile(
    const unsigned short* __restrict__ G, int gbase, int limit,
    int K, int k0, unsigned short (*S)[BK], int tid, bool full)
{
#if USE_ASYNC_LDS
    if (full) {
        #pragma unroll
        for (int q = 0; q < 2; ++q) {
            int i  = tid + q * GEMM_THREADS;   // 0..511 oct-chunks of 8 bf16
            int r  = i >> 2;
            int kk = (i & 3) << 3;
            __builtin_amdgcn_global_load_async_to_lds_b128(
                (v4i*)const_cast<unsigned short*>(
                    &G[(size_t)(gbase + r) * K + (size_t)(k0 + kk)]),
                (v4i*)&S[r][kk],
                0, 0);
        }
        return;
    }
#endif
    #pragma unroll
    for (int q = 0; q < 2; ++q) {
        int i  = tid + q * GEMM_THREADS;
        int r  = i >> 2;
        int kk = (i & 3) << 3;
        uint4 v = make_uint4(0u, 0u, 0u, 0u);
        if (full || (gbase + r) < limit)
            v = *reinterpret_cast<const uint4*>(
                    &G[(size_t)(gbase + r) * K + (size_t)(k0 + kk)]);
        *reinterpret_cast<uint4*>(&S[r][kk]) = v;
    }
}

// ---------------- GEMM: C[M,Fout] = A[M,K] * Bt[Fout,K]^T  (bf16 in, f32 out) ----------------
// A: bf16 row-major [M,K]; Bt: bf16 [Fout,K] (pre-transposed weights). K % 32 == 0.
__global__ __launch_bounds__(GEMM_THREADS)
void gcn_gemm_bf16(const unsigned short* __restrict__ A,
                   const unsigned short* __restrict__ Bt,
                   float* __restrict__ C, int M, int K, int Fout)
{
    __shared__ __align__(16) unsigned short As[2][BM][BK]; // double-buffered [row][k]
    __shared__ __align__(16) unsigned short Bs[2][BN][BK]; // double-buffered [col][k]

    const int tid  = threadIdx.x;
    const int lane = tid & 31;
    const int wave = tid >> 5;
    const int half = lane >> 4;   // 0 or 1
    const int l16  = lane & 15;

    const int mw = wave & 3;      // wave row group: 32 rows
    const int nw = wave >> 2;     // wave col group: 64 cols
    const int mbase = mw << 5;
    const int nbase = nw << 6;

    const int row0 = blockIdx.x * BM;
    const int col0 = blockIdx.y * BN;

    const bool a_full = (row0 + BM) <= M;
    const bool b_full = (col0 + BN) <= Fout;

    v8f acc[2][4] = {};

    const int ktiles = K / BK;

    // prologue: stage tile 0 into buffer 0
    stage_tile(A,  row0, M,    K, 0, As[0], tid, a_full);
    stage_tile(Bt, col0, Fout, K, 0, Bs[0], tid, b_full);

    for (int kt = 0; kt < ktiles; ++kt) {
#if USE_ASYNC_LDS
        __builtin_amdgcn_s_wait_asynccnt(0);
#endif
        __syncthreads();   // staged tile kt visible to all waves; prev buffer free

        const int cur = kt & 1;

        // stage tile kt+1 into the other buffer, overlapped with this tile's WMMAs
        if (kt + 1 < ktiles) {
            const int k0n = (kt + 1) * BK;
            stage_tile(A,  row0, M,    K, k0n, As[cur ^ 1], tid, a_full);
            stage_tile(Bt, col0, Fout, K, k0n, Bs[cur ^ 1], tid, b_full);
            // pull the tile after next toward L2
            if (kt + 2 < ktiles && a_full)
                __builtin_prefetch(&A[(size_t)(row0 + (tid >> 1)) * K
                                      + (size_t)(k0n + BK) + (size_t)((tid & 1) * 16)],
                                   0, 1);
        }

        // ---- A fragments (2 per wave), B fragments (4 per wave), 8 WMMAs ----
        const int m0 = mbase + l16;
        v16bf a0 = load_frag(&As[cur][m0][half * 8],      &As[cur][m0][16 + half * 8]);
        v16bf a1 = load_frag(&As[cur][m0 + 16][half * 8], &As[cur][m0 + 16][16 + half * 8]);

        #pragma unroll
        for (int ni = 0; ni < 4; ++ni) {
            const int n = nbase + (ni << 4) + l16;
            v16bf b = load_frag(&Bs[cur][n][half * 8], &Bs[cur][n][16 + half * 8]);
            acc[0][ni] = __builtin_amdgcn_wmma_f32_16x16x32_bf16(
                             false, a0, false, b, (short)0, acc[0][ni], false, false);
            acc[1][ni] = __builtin_amdgcn_wmma_f32_16x16x32_bf16(
                             false, a1, false, b, (short)0, acc[1][ni], false, false);
        }
        // no trailing barrier: WAR on As/Bs[cur] is protected by the next
        // iteration's barrier (all fragment data lands in VGPRs before WMMA issue)
    }

    // ---- store: C/D layout — lane = N, VGPR v = row half*8+v ----
    #pragma unroll
    for (int mi = 0; mi < 2; ++mi) {
        #pragma unroll
        for (int ni = 0; ni < 4; ++ni) {
            int col = col0 + nbase + (ni << 4) + l16;
            if (col >= Fout) continue;
            #pragma unroll
            for (int v = 0; v < 8; ++v) {
                int row = row0 + mbase + (mi << 4) + half * 8 + v;
                if (row < M)
                    C[(size_t)row * Fout + col] = acc[mi][ni][v];
            }
        }
    }
}

// ---------------- utility kernels ----------------
__global__ void gcn_zero(float* __restrict__ p, size_t n) {
    size_t i = (size_t)blockIdx.x * blockDim.x + threadIdx.x;
    if (i < n) p[i] = 0.f;
}

__global__ void gcn_cvt_bf16(const float* __restrict__ in, unsigned short* __restrict__ out,
                             size_t n) {
    size_t i = (size_t)blockIdx.x * blockDim.x + threadIdx.x;
    if (i < n) out[i] = f2bf(in[i]);
}

// W[K,Fout] fp32 -> Wt[Fout,K] bf16 (transpose + convert)
__global__ void gcn_cvt_w(const float* __restrict__ W, unsigned short* __restrict__ Wt,
                          int K, int Fout) {
    int i = blockIdx.x * blockDim.x + threadIdx.x;
    int total = K * Fout;
    if (i >= total) return;
    int n = i / K;
    int k = i - n * K;
    Wt[(size_t)n * K + k] = f2bf(W[(size_t)k * Fout + n]);
}

__global__ void gcn_degree(const int* __restrict__ src, const int* __restrict__ dst,
                           float* __restrict__ dsrc, float* __restrict__ ddst, int E) {
    int i = blockIdx.x * blockDim.x + threadIdx.x;
    if (i < E) {
        atomicAdd(&dsrc[src[i]], 1.f);
        atomicAdd(&ddst[dst[i]], 1.f);
    }
}

__global__ void gcn_invsqrt(float* __restrict__ d, int n) {
    int i = blockIdx.x * blockDim.x + threadIdx.x;
    if (i < n) {
        float v = d[i];
        d[i] = (v > 0.f) ? rsqrtf(fmaxf(v, 1.f)) : 0.f;
    }
}

// scatter-add: agg[dst[e], :] += hw[src[e], :] * c_src[src[e]]  (L2-resident atomics)
__global__ void gcn_edge_agg(const float* __restrict__ hw,
                             const int* __restrict__ src, const int* __restrict__ dst,
                             const float* __restrict__ c_src,
                             float* __restrict__ agg, int E, int F) {
    int e = blockIdx.x;
    if (e >= E) return;
    int s = src[e], d = dst[e];
    float cs = c_src[s];
    const float* hs = hw  + (size_t)s * F;
    float*       ad = agg + (size_t)d * F;
    for (int f = threadIdx.x; f < F; f += blockDim.x)
        atomicAdd(&ad[f], hs[f] * cs);
}

// out_bf16 = bf16(relu(agg * c_dst + b)) : feeds next layer's GEMM directly
__global__ void gcn_finalize_bf16(const float* __restrict__ agg, const float* __restrict__ c_dst,
                                  const float* __restrict__ bias, unsigned short* __restrict__ out,
                                  int Nn, int F) {
    size_t i = (size_t)blockIdx.x * blockDim.x + threadIdx.x;
    size_t total = (size_t)Nn * F;
    if (i >= total) return;
    int n = (int)(i / F);
    int f = (int)(i % F);
    float v = fmaxf(agg[i] * c_dst[n] + bias[f], 0.f);
    out[i] = f2bf(v);
}

// out_f32 = agg * c_dst + b   (final layer, no activation)
__global__ void gcn_finalize_f32(const float* __restrict__ agg, const float* __restrict__ c_dst,
                                 const float* __restrict__ bias, float* __restrict__ out,
                                 int Nn, int F) {
    size_t i = (size_t)blockIdx.x * blockDim.x + threadIdx.x;
    size_t total = (size_t)Nn * F;
    if (i >= total) return;
    int n = (int)(i / F);
    int f = (int)(i % F);
    out[i] = agg[i] * c_dst[n] + bias[f];
}

// ---------------- one GCN layer ----------------
static void run_layer(const unsigned short* Hbf_in, const float* Wf, const float* bias,
                      const int* src, const int* dst,
                      const float* c_src, const float* c_dst,
                      unsigned short* Wt, float* hw, float* agg,
                      unsigned short* Hbf_out, float* outf,
                      int Nn, int E, int K, int Fout, hipStream_t stream)
{
    int wtotal = K * Fout;
    gcn_cvt_w<<<(wtotal + 255) / 256, 256, 0, stream>>>(Wf, Wt, K, Fout);

    dim3 g((Nn + BM - 1) / BM, (Fout + BN - 1) / BN);
    gcn_gemm_bf16<<<g, GEMM_THREADS, 0, stream>>>(Hbf_in, Wt, hw, Nn, K, Fout);

    size_t cnt = (size_t)Nn * Fout;
    int zb = (int)((cnt + 255) / 256);
    gcn_zero<<<zb, 256, 0, stream>>>(agg, cnt);

    int bt = (Fout >= 256) ? 256 : 64;
    gcn_edge_agg<<<E, bt, 0, stream>>>(hw, src, dst, c_src, agg, E, Fout);

    if (Hbf_out)
        gcn_finalize_bf16<<<zb, 256, 0, stream>>>(agg, c_dst, bias, Hbf_out, Nn, Fout);
    else
        gcn_finalize_f32<<<zb, 256, 0, stream>>>(agg, c_dst, bias, outf, Nn, Fout);
}

extern "C" void kernel_launch(void* const* d_in, const int* in_sizes, int n_in,
                              void* d_out, int out_size, void* d_ws, size_t ws_size,
                              hipStream_t stream)
{
    const float* x   = (const float*)d_in[0];
    const int*   src = (const int*)d_in[1];
    const int*   dst = (const int*)d_in[2];
    const float* W1  = (const float*)d_in[3];  const float* b1 = (const float*)d_in[4];
    const float* W2  = (const float*)d_in[5];  const float* b2 = (const float*)d_in[6];
    const float* W3  = (const float*)d_in[7];  const float* b3 = (const float*)d_in[8];
    const float* W4  = (const float*)d_in[9];  const float* b4 = (const float*)d_in[10];

    const int IN_F = 256, HID = 512;
    const int Nn = in_sizes[0] / IN_F;     // 50000
    const int E  = in_sizes[1];            // 800000
    const int NC = in_sizes[10];           // 40

    float* out = (float*)d_out;

    // workspace layout
    float* c_src = (float*)d_ws;                           // [Nn]
    float* c_dst = c_src + Nn;                             // [Nn]
    float* hw    = c_dst + Nn;                             // [Nn,512] f32 post-GEMM
    float* agg   = hw   + (size_t)Nn * HID;                // [Nn,512] f32 scatter target
    unsigned short* Hbf = (unsigned short*)(agg + (size_t)Nn * HID); // [Nn,512] bf16 acts
    unsigned short* Wt  = Hbf + (size_t)Nn * HID;          // [512,512] bf16 transposed W

    // degree normalizers
    gcn_zero<<<(2 * Nn + 255) / 256, 256, 0, stream>>>(c_src, (size_t)2 * Nn);
    gcn_degree<<<(E + 255) / 256, 256, 0, stream>>>(src, dst, c_src, c_dst, E);
    gcn_invsqrt<<<(2 * Nn + 255) / 256, 256, 0, stream>>>(c_src, 2 * Nn);

    // x -> bf16 activations
    size_t xcnt = (size_t)Nn * IN_F;
    gcn_cvt_bf16<<<(int)((xcnt + 255) / 256), 256, 0, stream>>>(x, Hbf, xcnt);

    run_layer(Hbf, W1, b1, src, dst, c_src, c_dst, Wt, hw, agg, Hbf, nullptr,
              Nn, E, IN_F, HID, stream);
    run_layer(Hbf, W2, b2, src, dst, c_src, c_dst, Wt, hw, agg, Hbf, nullptr,
              Nn, E, HID, HID, stream);
    run_layer(Hbf, W3, b3, src, dst, c_src, c_dst, Wt, hw, agg, Hbf, nullptr,
              Nn, E, HID, HID, stream);
    run_layer(Hbf, W4, b4, src, dst, c_src, c_dst, Wt, hw, agg, nullptr, out,
              Nn, E, HID, NC, stream);

    (void)n_in; (void)out_size; (void)ws_size;
}
